// ResidualDCM_66563403153840
// MI455X (gfx1250) — matrix operand order repaired and checked by reading, not compiled, not run on hardware
//
#include <hip/hip_runtime.h>

// ======================================================================
// ResidualDCM on MI455X (gfx1250): persistent single-WGP RK4 integrator.
//  - prep_weights: swizzle W1/[A|C]/Bstack and W2 into bf16 WMMA A-fragment
//    tiles in d_ws (each 16x32 tile = 32 lanes x 16 bf16 = 1KB, lane-major).
//  - residual_dcm: one 1024-thread workgroup (32 wave32), state in LDS,
//    2047 RK4 steps x 4 dyn() evals, each dyn = two WMMA mat-vec phases
//    (v_wmma_f32_16x16x32_bf16) + elementwise hemodynamics.
//    Phase-1 vector fragments are hoisted into registers and reused
//    across all row-tiles (no DS/EXEC traffic in the steady-state loop).
// Scratch requirement: d_ws >= (P_ELEMS + W2_ELEMS) * 2 bytes  (~2.4 MB).
// ======================================================================

constexpr int cL = 256, cM = 8, cH = 1024, cT = 2048;
constexpr int cS = 5 * cL;                  // 1280-float state
constexpr int KPAD  = 288;                  // (L+M)=264 padded to 9*32
constexpr int NKT1  = KPAD / 32;            // 9 K-tiles, phase 1
constexpr int NR1   = cH + cL + cM * cL;    // 3328 stacked rows: W1 | [A|C] | Bstack
constexpr int NRT1  = NR1 / 16;             // 208 row-tiles
constexpr int NKT2  = cH / 32;              // 32 K-tiles, phase 2 (W2)
constexpr int NRT2  = cL / 16;              // 16 row-tiles
constexpr int TILE_E   = 512;               // 32 lanes * 16 bf16 per tile
constexpr int P_ELEMS  = NRT1 * NKT1 * TILE_E;   // 958464
constexpr int W2_ELEMS = NRT2 * NKT2 * TILE_E;   // 262144

constexpr float cKAPPA = 0.65f, cGAMMA = 0.41f, cTAU = 0.98f;
constexpr float cRHO = 0.34f, cV0 = 0.04f, cALPHA = 0.1f;
constexpr float cINVH = 3.125f;                  // 1/0.32
constexpr float cLOG1MRHO = -0.41551544396f;     // ln(1-0.34)
constexpr float cK1B = 7.0f * cRHO, cK2B = 2.0f, cK3B = 2.0f * cRHO - 0.2f;

typedef __attribute__((ext_vector_type(16))) __bf16 v16bf;
typedef __attribute__((ext_vector_type(8)))  float  v8f;

__device__ __forceinline__ __bf16 f2bf(float f) {
  unsigned u = __builtin_bit_cast(unsigned, f);
  unsigned r = (u + 0x7FFFu + ((u >> 16) & 1u)) >> 16;   // RNE
  return __builtin_bit_cast(__bf16, (unsigned short)r);
}

__device__ __forceinline__ v8f wmma_bf16(v16bf a, v16bf b, v8f c) {
  // (neg_a, A, neg_b, B, c_mod, C, reuse_a, reuse_b)
  return __builtin_amdgcn_wmma_f32_16x16x32_bf16(false, a, false, b,
                                                 (short)0, c, false, false);
}

// B-operand (32x16 bf16, K x N): matvec uses only column N==0, held by
// lanes 0 (K = kbase..kbase+15) and 16 (K = kbase+16..kbase+31).
__device__ __forceinline__ v16bf vec_frag(const __bf16* vbase, int kbase, int lane) {
  v16bf z;
#pragma unroll
  for (int i = 0; i < 16; ++i) z[i] = __builtin_bit_cast(__bf16, (unsigned short)0);
  if ((lane & 15) == 0)
    z = *(const v16bf*)(vbase + kbase + ((lane >> 4) << 4));   // 32B aligned LDS load
  return z;
}

// ---------------------------------------------------------------------
// Prep: swizzle f32 weights -> bf16 WMMA A-fragment tiles in workspace.
// A-fragment lane map (16-bit A 16x32): row = lane&15; half = lane>>4;
// elem e -> K = e + (half<<4).
// ---------------------------------------------------------------------
__global__ void prep_weights(const float* __restrict__ A, const float* __restrict__ B,
                             const float* __restrict__ C, const float* __restrict__ W1,
                             const float* __restrict__ W2, __bf16* __restrict__ ws) {
  const int total = P_ELEMS + W2_ELEMS;
  for (int id = blockIdx.x * blockDim.x + threadIdx.x; id < total;
       id += gridDim.x * blockDim.x) {
    const bool isP = id < P_ELEMS;
    const int lid  = isP ? id : id - P_ELEMS;
    const int e    = lid & 15;
    const int lane = (lid >> 4) & 31;
    const int tile = lid >> 9;
    const int half = lane >> 4;
    const int kof  = e + (half << 4);
    float val = 0.0f;
    if (isP) {
      const int rt = tile / NKT1, kt = tile % NKT1;
      const int row = rt * 16 + (lane & 15);
      const int k   = kt * 32 + kof;
      if (row < cH) {                         // W1 (1024 x 264)
        if (k < cL + cM) val = W1[row * (cL + cM) + k];
      } else if (row < cH + cL) {             // G = [A | C] (256 x 264)
        const int r = row - cH;
        if (k < cL)           val = A[r * cL + k];
        else if (k < cL + cM) val = C[r * cM + (k - cL)];
      } else {                                // Bstack (2048 x 256), r2 = m*256+i
        const int r2 = row - (cH + cL);
        if (k < cL) val = B[(size_t)r2 * cL + k];
      }
    } else {
      const int rt = tile / NKT2, kt = tile % NKT2;
      const int row = rt * 16 + (lane & 15);
      const int k   = kt * 32 + kof;
      val = W2[(size_t)row * cH + k];         // W2 (256 x 1024), K always in range
    }
    ws[id] = f2bf(val);
  }
}

// ---------------------------------------------------------------------
// Persistent integrator: one workgroup, 32 wave32.
// ---------------------------------------------------------------------
__global__ __launch_bounds__(1024, 1)
void residual_dcm(const float* __restrict__ gb1, const float* __restrict__ gb2,
                  const float* __restrict__ u_seq, const float* __restrict__ t_eval,
                  const __bf16* __restrict__ ws, float* __restrict__ out) {
  __shared__ float st[cS], cur[cS], racc[cS], kd[cS];
  __shared__ float p1out[NR1];          // [0,1024): W1@zu | [1024,1280): A@z+C@u | [1280,3328): Bstack@z
  __shared__ float p2out[2][cL];        // K-split partials of W2@h
  __shared__ float dzb[cL];
  __shared__ float b1l[cH], b2l[cL];
  __shared__ float u3[3][cM];           // u0, um, u1
  __shared__ float sh_h[1];
  __shared__ __attribute__((aligned(32))) __bf16 zu_bf[KPAD];
  __shared__ __attribute__((aligned(32))) __bf16 h_bf[cH];

  const int tid  = threadIdx.x;
  const int lane = tid & 31;
  const int wave = tid >> 5;
  const __bf16* wsP  = ws;
  const __bf16* wsW2 = ws + P_ELEMS;

  // ---- init: s0 = [0,0, 1,1,1] blocks; emit S row 0 and Y row 0 ----
  for (int i = tid; i < cS; i += 1024) {
    const float v = (i < 2 * cL) ? 0.0f : 1.0f;
    st[i] = v;
    out[i] = v;                                        // S[0]
  }
  if (tid < cL) out[(size_t)cT * cS + tid] = 0.0f;     // Y[0] (v=q=1 -> 0)
  for (int i = tid; i < cH; i += 1024) b1l[i] = gb1[i];
  if (tid < cL) b2l[tid] = gb2[tid];
  if (tid >= cL + cM && tid < KPAD) zu_bf[tid] = f2bf(0.0f);   // static K padding
  __syncthreads();

  for (int t = 0; t < cT - 1; ++t) {
    // step start: u triplet + h; cur = state
    if (tid < 3 * cM) u3[tid / cM][tid % cM] = u_seq[(size_t)(2 * t + tid / cM) * cM + (tid % cM)];
    if (tid == 31) sh_h[0] = t_eval[t + 1] - t_eval[t];
    for (int i = tid; i < cS; i += 1024) cur[i] = st[i];
    __syncthreads();

#pragma unroll 1
    for (int sub = 0; sub < 4; ++sub) {
      const float* u_t = (sub == 0) ? u3[0] : (sub == 3 ? u3[2] : u3[1]);

      // ---- dyn(cur, u_t) -> kd ----
      // pass A: zu = bf16([z ; u ; pad])
      if (tid < cL)             zu_bf[tid] = f2bf(cur[tid]);
      else if (tid < cL + cM)   zu_bf[tid] = f2bf(u_t[tid - cL]);
      __syncthreads();

      // phase 1: [W1; A|C; Bstack] @ zu  (208 row-tiles x 9 K-tiles)
      {
        // Hoist all 9 vector fragments into registers; they are invariant
        // across row-tiles, so the inner loop is pure global-load + wmma.
        v16bf bq[NKT1];
#pragma unroll
        for (int kt = 0; kt < NKT1; ++kt) bq[kt] = vec_frag(zu_bf, kt * 32, lane);

        for (int rt = wave; rt < NRT1; rt += 32) {
          v8f c8 = {};
          const __bf16* tb = wsP + (size_t)rt * (NKT1 * TILE_E) + lane * 16;
          if (rt + 32 < NRT1)
            __builtin_prefetch(tb + (size_t)32 * NKT1 * TILE_E, 0, 3);
#pragma unroll
          for (int kt = 0; kt < NKT1; ++kt) {
            v16bf a = *(const v16bf*)(tb + (size_t)kt * TILE_E);
            c8 = wmma_bf16(a, bq[kt], c8);
          }
          if ((lane & 15) == 0) {                       // D column 0: lanes 0 / 16
            const int mb = rt * 16 + ((lane >> 4) << 3);
#pragma unroll
            for (int g = 0; g < 8; ++g) p1out[mb + g] = c8[g];
          }
        }
      }
      __syncthreads();

      // pass B: h = bf16(tanh(W1@zu + b1));  dzb_i = sum_m u_m * Bz[m,i]
      {
        h_bf[tid] = f2bf(tanhf(p1out[tid] + b1l[tid]));
        if (tid < cL) {
          float s = 0.0f;
#pragma unroll
          for (int m = 0; m < cM; ++m) s += u_t[m] * p1out[cH + cL + m * cL + tid];
          dzb[tid] = s;
        }
      }
      __syncthreads();

      // phase 2: W2 @ h  (16 row-tiles; 2 waves per row-tile split K 16+16)
      {
        const int rt = wave & 15;
        const int hk = wave >> 4;
        v8f c8 = {};
        const __bf16* tb = wsW2 + (size_t)(rt * NKT2 + hk * 16) * TILE_E + lane * 16;
#pragma unroll 4
        for (int kt = 0; kt < 16; ++kt) {
          v16bf a = *(const v16bf*)(tb + (size_t)kt * TILE_E);
          v16bf b = vec_frag(h_bf, (hk * 16 + kt) * 32, lane);
          c8 = wmma_bf16(a, b, c8);
        }
        if ((lane & 15) == 0) {
          const int mb = rt * 16 + ((lane >> 4) << 3);
#pragma unroll
          for (int g = 0; g < 8; ++g) p2out[hk][mb + g] = c8[g];
        }
      }
      __syncthreads();

      // pass C: assemble derivatives
      if (tid < cL) {
        const int i = tid;
        const float z  = cur[i];
        const float s_ = cur[cL + i];
        const float f_ = cur[2 * cL + i];
        const float v_ = cur[3 * cL + i];
        const float q_ = cur[4 * cL + i];
        const float res = p2out[0][i] + p2out[1][i] + b2l[i];
        const float dz  = p1out[cH + i] + dzb[i] + cALPHA * res;  // A@z + C@u + Bz.u + a*res
        const float ds  = z - cKAPPA * s_ - cGAMMA * (f_ - 1.0f);
        const float fv  = __powf(v_, cINVH);
        const float dv  = (f_ - fv) / cTAU;
        const float E   = 1.0f - __expf(cLOG1MRHO / f_);
        const float dq  = (f_ * E / cRHO - fv * q_ / v_) / cTAU;
        kd[i] = dz; kd[cL + i] = ds; kd[2 * cL + i] = s_;
        kd[3 * cL + i] = dv; kd[4 * cL + i] = dq;
      }
      __syncthreads();

      // RK4 combine
      {
        const float h = sh_h[0];
        if (sub < 3) {
          const float cf = (sub == 2) ? h : 0.5f * h;
          const float wf = (sub == 0) ? 1.0f : 2.0f;
          for (int i = tid; i < cS; i += 1024) {
            const float kv = kd[i];
            cur[i]  = st[i] + cf * kv;
            racc[i] = (sub == 0) ? kv : racc[i] + wf * kv;
          }
        } else {
          const float h6 = h * (1.0f / 6.0f);
          for (int i = tid; i < cS; i += 1024)
            st[i] += h6 * (racc[i] + kd[i]);
        }
      }
      __syncthreads();
    }

    // emit S[t+1] and Y[t+1]
    for (int i = tid; i < cS; i += 1024) out[(size_t)(t + 1) * cS + i] = st[i];
    if (tid < cL) {
      const float v_ = st[3 * cL + tid], q_ = st[4 * cL + tid];
      const float Y = cV0 * (cK1B * (1.0f - q_) + cK2B * (1.0f - q_ / v_) + cK3B * (1.0f - v_));
      out[(size_t)cT * cS + (size_t)(t + 1) * cL + tid] = Y;
    }
    __syncthreads();
  }
}

extern "C" void kernel_launch(void* const* d_in, const int* in_sizes, int n_in,
                              void* d_out, int out_size, void* d_ws, size_t ws_size,
                              hipStream_t stream) {
  (void)in_sizes; (void)n_in; (void)out_size; (void)ws_size;
  const float* A      = (const float*)d_in[0];
  const float* B      = (const float*)d_in[1];
  const float* C      = (const float*)d_in[2];
  const float* W1     = (const float*)d_in[3];
  const float* b1     = (const float*)d_in[4];
  const float* W2     = (const float*)d_in[5];
  const float* b2     = (const float*)d_in[6];
  const float* u_seq  = (const float*)d_in[7];
  const float* t_eval = (const float*)d_in[8];
  __bf16* ws = (__bf16*)d_ws;
  float* out = (float*)d_out;

  prep_weights<<<1024, 256, 0, stream>>>(A, B, C, W1, W2, ws);
  residual_dcm<<<1, 1024, 0, stream>>>(b1, b2, u_seq, t_eval, ws, out);
}
